// Model_39676907884573
// MI455X (gfx1250) — matrix operand order repaired and checked by reading, not compiled, not run on hardware
//
#include <hip/hip_runtime.h>
#include <hip/hip_bf16.h>

typedef __attribute__((ext_vector_type(16))) _Float16 v16h;
typedef __attribute__((ext_vector_type(8)))  float    v8f;

#define B_ 4
#define H_ 16
#define L_ 1024
#define D_ 64
#define WAVES_ 16
#define ROWS_PER_WG (WAVES_ * 16)
#define THREADS_ (WAVES_ * 32)
#define MT_ (L_ / 16)   // 64 m-tiles

__device__ __forceinline__ float elu_f(float x) {
    return x > 0.0f ? x : (__expf(x) - 1.0f);
}

// out[b,h,l,:] = drop_mask[b,h,l,m*] * elu(v[b,h,m*,:]),
// m* = argmax_m( elu(q_l).elu(k_m)/sqrt(D) + gumbel[l,m] )
extern "C" __global__ __launch_bounds__(THREADS_)
void gumbel_hard_attn(const float* __restrict__ q,
                      const float* __restrict__ k,
                      const float* __restrict__ v,
                      const float* __restrict__ gumbel,
                      const float* __restrict__ drop,
                      float* __restrict__ out)
{
    extern __shared__ _Float16 kl[];           // elu(k) as f16: 1024 x 64 = 128 KB

    const int tid  = threadIdx.x;
    const int lane = tid & 31;
    const int wave = tid >> 5;
    const int g    = lane >> 4;                // 16-lane group within wave
    const int lr   = lane & 15;

    const int bh = blockIdx.y;                 // fused (b,h)
    const int l0 = blockIdx.x * ROWS_PER_WG + wave * 16;   // wave's 16 q rows

    // ---- stage elu(k) as f16 into LDS (shared by all 16 waves) ----
    const float* kbh = k + (size_t)bh * L_ * D_;
    for (int idx = tid; idx < L_ * D_; idx += THREADS_)
        kl[idx] = (_Float16)elu_f(kbh[idx]);

    // ---- build A fragments (16x32 f16 per K-chunk), contiguous-K layout used here (same as B):
    //      lane m-row = lane&15 ; contiguous K = base..base+15,
    //      base = (lane>>4)*16 ; chunk1 adds +32.
    const float* qrow = q + ((size_t)bh * L_ + (size_t)(l0 + lr)) * D_;
    v16h a0, a1;
    {
        const int base = g * 16;
        #pragma unroll
        for (int j = 0; j < 8; ++j) {
            a0[j]     = (_Float16)elu_f(qrow[base + j]);
            a0[8 + j] = (_Float16)elu_f(qrow[base + 8 + j]);
            a1[j]     = (_Float16)elu_f(qrow[base + 32 + j]);
            a1[8 + j] = (_Float16)elu_f(qrow[base + 40 + j]);
        }
    }

    __syncthreads();

    float best[8];
    int   bestm[8];
    #pragma unroll
    for (int r = 0; r < 8; ++r) { best[r] = -3.4e38f; bestm[r] = 0; }

    const float* gbase = gumbel + ((size_t)bh * L_ + (size_t)(l0 + g * 8)) * L_ + lr;

    // B fragments (32x16 f16): lane n-col = lane&15, contiguous K halves,
    // group0 -> K 0..15, group1 -> K 16..31 (chunk1 adds +32). 32B aligned.
    auto loadB = [&](int mt, v16h& b0, v16h& b1) {
        const _Float16* krow = kl + (size_t)(mt * 16 + lr) * D_;
        b0 = *(const v16h*)(krow + g * 16);
        b1 = *(const v16h*)(krow + 32 + g * 16);
    };
    // gumbel: streaming, single use -> nontemporal so the 256 MB stream
    // doesn't wash k/v out of L2.
    auto loadG = [&](int mt, float (&gg)[8]) {
        const float* gp = gbase + mt * 16;
        #pragma unroll
        for (int r = 0; r < 8; ++r)
            gg[r] = __builtin_nontemporal_load(gp + (size_t)r * L_);
    };
    // two WMMAs + argmax update against already-resident gumbel values
    auto process = [&](int mt, const v16h& b0, const v16h& b1, const float (&gg)[8]) {
        v8f c = {};
        c = __builtin_amdgcn_wmma_f32_16x16x32_f16(false, a0, false, b0,
                                                   (short)0, c, false, false);
        c = __builtin_amdgcn_wmma_f32_16x16x32_f16(false, a1, false, b1,
                                                   (short)0, c, false, false);
        const int m = mt * 16 + lr;
        #pragma unroll
        for (int r = 0; r < 8; ++r) {
            float val = c[r] * 0.125f + gg[r];
            if (val > best[r]) { best[r] = val; bestm[r] = m; }  // keep first max
        }
    };

    // ---- software-pipelined main loop, ping-pong buffers (no reg moves) ----
    v16h b0a, b1a, b0b, b1b;
    float ga[8], gb[8];
    loadB(0, b0a, b1a);
    loadG(0, ga);

    for (int mt = 0; mt < MT_; mt += 2) {
        loadB(mt + 1, b0b, b1b);              // prefetch tile mt+1 -> buf B
        loadG(mt + 1, gb);
        process(mt, b0a, b1a, ga);

        loadB((mt + 2) & (MT_ - 1), b0a, b1a); // prefetch tile mt+2 -> buf A
        loadG((mt + 2) & (MT_ - 1), ga);       // (wraps harmlessly at the end)
        process(mt + 1, b0b, b1b, gb);
    }

    // ---- cross-lane argmax within each 16-lane group (xor 1,2,4,8) ----
    #pragma unroll
    for (int r = 0; r < 8; ++r) {
        float bv = best[r];
        int   bm = bestm[r];
        #pragma unroll
        for (int off = 1; off <= 8; off <<= 1) {
            float ov = __shfl_xor(bv, off, 32);
            int   om = __shfl_xor(bm, off, 32);
            if (ov > bv || (ov == bv && om < bm)) { bv = ov; bm = om; }
        }
        bestm[r] = bm;
    }

    // ---- out[l,:] = drop[l,m*] * elu(v[m*,:]); 16 lanes x float4 = 64 floats ----
    #pragma unroll
    for (int r = 0; r < 8; ++r) {
        const int l     = l0 + g * 8 + r;
        const int mstar = bestm[r];
        const float dm  = drop[((size_t)bh * L_ + l) * L_ + mstar];
        const float4 vv = *(const float4*)(v + ((size_t)bh * L_ + mstar) * D_ + lr * 4);
        float4 o;
        o.x = elu_f(vv.x) * dm;
        o.y = elu_f(vv.y) * dm;
        o.z = elu_f(vv.z) * dm;
        o.w = elu_f(vv.w) * dm;
        *(float4*)(out + ((size_t)bh * L_ + l) * D_ + lr * 4) = o;
    }
}

extern "C" void kernel_launch(void* const* d_in, const int* in_sizes, int n_in,
                              void* d_out, int out_size, void* d_ws, size_t ws_size,
                              hipStream_t stream) {
    const float* q  = (const float*)d_in[0];
    const float* k  = (const float*)d_in[1];
    const float* v  = (const float*)d_in[2];
    const float* gn = (const float*)d_in[3];
    const float* dm = (const float*)d_in[4];
    float* out      = (float*)d_out;

    dim3 grid(L_ / ROWS_PER_WG, B_ * H_);     // (4, 64)
    dim3 block(THREADS_);                     // 512 threads = 16 wave32
    size_t shmem = (size_t)L_ * D_ * sizeof(_Float16);  // 128 KB elu(k) f16
    gumbel_hard_attn<<<grid, block, shmem, stream>>>(q, k, v, gn, dm, out);
}
